// Model_70617852281169
// MI455X (gfx1250) — compile-verified, matrix operand order checked by reading, not run on hardware
//
#include <hip/hip_runtime.h>
#include <cstdint>
#include <cstddef>

// ---------------------------------------------------------------------------
// MI455X (gfx1250, wave32) implementation.
//
// Heavy GEMMs use v_wmma_f32_16x16x32_f16 (f16 A/B, f32 accumulate).
// A is activations [M, Kp] row-major f16; B is the weight matrix w [N, K]
// row-major f16 (padded to [Np, Kp]).  out[m,n] = sum_k A[m,k] * w[n,k],
// which maps to WMMA D = A x B with B[k,n] = w[n,k]; both fragments are
// K-contiguous per lane so both load with plain 16-byte global loads.
//
// Wave tile: 64x32 (4x2 WMMA tiles) -> 8 WMMAs per 6 fragment loads per
// K-step, maximizing WMMA issue density vs VMEM slots.  Block: 8 waves
// arranged 2(M) x 4(N) -> 128x128 block tile.
//
// Workspace layout (halves unless noted), ~162 MB total:
//   f16 weights (9 layers, K padded to 32, N padded to 128-tiles for finals)
//   ACT_A, ACT_B : ping-pong f16 activations [8192, 2048]
//   F32 scratch  : pre-groupnorm GEMM output  [8192, 2048] f32
// ---------------------------------------------------------------------------

typedef __attribute__((ext_vector_type(16))) _Float16 v16h;
typedef __attribute__((ext_vector_type(8)))  float    v8f;

#define B_ROWS 8192

// ------------------------- WMMA fragment loader ----------------------------
union FragU { uint4 q[2]; v16h h; };

// base must point at (row, k + 8*(lane>=16)) and be 16B-aligned.
// Loads K offsets {0..7} and {16..23} relative to base, matching the
// CDNA5 16-bit A/B fragment layout (CDNA5 ISA, 16x32 A-matrix table).
__device__ __forceinline__ v16h load_frag(const _Float16* base) {
  FragU f;
  f.q[0] = *reinterpret_cast<const uint4*>(base);
  f.q[1] = *reinterpret_cast<const uint4*>(base + 16);
  return f.h;
}

// ------------------------------ GEMM kernel --------------------------------
// Block: 256 threads = 8 waves, arranged 2 (M) x 4 (N).
// Each wave computes a 64x32 output tile (4x2 WMMA tiles) -> block 128x128.
__global__ __launch_bounds__(256)
void gemm_wmma_kernel(const _Float16* __restrict__ A,   // [M, Kp] f16
                      const _Float16* __restrict__ W,   // [Np, Kp] f16
                      const float*    __restrict__ bias,// [>= Nlim] f32
                      float*          __restrict__ out, // [M, out_stride] f32
                      int Kp, int out_stride, int Nlim)
{
  const int lane  = threadIdx.x & 31;
  const int wave  = threadIdx.x >> 5;
  const int wm    = wave & 1;                 // 0..1  (M)
  const int wn    = wave >> 1;                // 0..3  (N)
  const int m_w   = blockIdx.x * 128 + wm * 64;
  const int n_w   = blockIdx.y * 128 + wn * 32;
  const int half8 = (lane >> 4) << 3;         // 0 for lanes 0-15, 8 for 16-31
  const int lm    = lane & 15;

  const _Float16* pa0 = A + (size_t)(m_w + lm)      * Kp + half8;
  const _Float16* pa1 = A + (size_t)(m_w + 16 + lm) * Kp + half8;
  const _Float16* pa2 = A + (size_t)(m_w + 32 + lm) * Kp + half8;
  const _Float16* pa3 = A + (size_t)(m_w + 48 + lm) * Kp + half8;
  const _Float16* pb0 = W + (size_t)(n_w + lm)      * Kp + half8;
  const _Float16* pb1 = W + (size_t)(n_w + 16 + lm) * Kp + half8;

  v8f acc00 = {}; v8f acc01 = {};
  v8f acc10 = {}; v8f acc11 = {};
  v8f acc20 = {}; v8f acc21 = {};
  v8f acc30 = {}; v8f acc31 = {};

  for (int k = 0; k < Kp; k += 32) {
    v16h b0 = load_frag(pb0 + k);
    v16h b1 = load_frag(pb1 + k);
    v16h a0 = load_frag(pa0 + k);
    v16h a1 = load_frag(pa1 + k);
    v16h a2 = load_frag(pa2 + k);
    v16h a3 = load_frag(pa3 + k);
    acc00 = __builtin_amdgcn_wmma_f32_16x16x32_f16(false, a0, false, b0,
                                                   (short)0, acc00, false, false);
    acc01 = __builtin_amdgcn_wmma_f32_16x16x32_f16(false, a0, false, b1,
                                                   (short)0, acc01, false, false);
    acc10 = __builtin_amdgcn_wmma_f32_16x16x32_f16(false, a1, false, b0,
                                                   (short)0, acc10, false, false);
    acc11 = __builtin_amdgcn_wmma_f32_16x16x32_f16(false, a1, false, b1,
                                                   (short)0, acc11, false, false);
    acc20 = __builtin_amdgcn_wmma_f32_16x16x32_f16(false, a2, false, b0,
                                                   (short)0, acc20, false, false);
    acc21 = __builtin_amdgcn_wmma_f32_16x16x32_f16(false, a2, false, b1,
                                                   (short)0, acc21, false, false);
    acc30 = __builtin_amdgcn_wmma_f32_16x16x32_f16(false, a3, false, b0,
                                                   (short)0, acc30, false, false);
    acc31 = __builtin_amdgcn_wmma_f32_16x16x32_f16(false, a3, false, b1,
                                                   (short)0, acc31, false, false);
  }

  // C/D layout: VGPR v holds row (v + 8*(lane>=16)), col = lane&15 of tile.
  const int col0  = n_w + lm;
  const int col1  = col0 + 16;
  const int rBase = m_w + half8;
  const float bv0 = (col0 < Nlim) ? bias[col0] : 0.0f;
  const float bv1 = (col1 < Nlim) ? bias[col1] : 0.0f;

  v8f acc_t0[4] = { acc00, acc10, acc20, acc30 };
  v8f acc_t1[4] = { acc01, acc11, acc21, acc31 };

#pragma unroll
  for (int t = 0; t < 4; ++t) {
#pragma unroll
    for (int v = 0; v < 8; ++v) {
      const int r = rBase + t * 16 + v;
      if (col0 < Nlim) out[(size_t)r * out_stride + col0] = acc_t0[t][v] + bv0;
      if (col1 < Nlim) out[(size_t)r * out_stride + col1] = acc_t1[t][v] + bv1;
    }
  }
}

// ------------------- GroupNorm (32 groups) + ReLU + cvt f16 ----------------
// One wave handles one (row, group).  C is 1024 or 2048 -> group size 32/64.
__global__ __launch_bounds__(256)
void gn_relu_cvt_kernel(const float* __restrict__ x,   // [B, C] f32
                        const float* __restrict__ gw,
                        const float* __restrict__ gb,
                        _Float16*    __restrict__ y,   // [B, C] f16
                        int C)
{
  const int lane = threadIdx.x & 31;
  const int wid  = (int)((blockIdx.x * blockDim.x + threadIdx.x) >> 5);
  const int row  = wid >> 5;       // / 32 groups
  const int grp  = wid & 31;
  const int Cg   = C >> 5;         // 32 or 64
  const size_t base = (size_t)row * C + grp * Cg;

  float v0 = x[base + lane];
  float v1 = (Cg == 64) ? x[base + 32 + lane] : 0.0f;
  float s  = v0 + v1;
  float s2 = v0 * v0 + v1 * v1;
#pragma unroll
  for (int m = 16; m >= 1; m >>= 1) {
    s  += __shfl_xor(s,  m, 32);
    s2 += __shfl_xor(s2, m, 32);
  }
  const float inv_n = 1.0f / (float)Cg;
  const float mean  = s * inv_n;
  const float var   = fmaxf(s2 * inv_n - mean * mean, 0.0f);
  const float rs    = rsqrtf(var + 1e-3f);

  const int c0 = grp * Cg + lane;
  y[base + lane] = (_Float16)fmaxf((v0 - mean) * rs * gw[c0] + gb[c0], 0.0f);
  if (Cg == 64) {
    const int c1 = c0 + 32;
    y[base + 32 + lane] = (_Float16)fmaxf((v1 - mean) * rs * gw[c1] + gb[c1], 0.0f);
  }
}

// ----------------------- weight f32 -> padded f16 --------------------------
__global__ void cvt_w_kernel(const float* __restrict__ w, _Float16* __restrict__ o,
                             int N, int K, int Np, int Kp)
{
  const int idx = blockIdx.x * blockDim.x + threadIdx.x;
  if (idx >= Np * Kp) return;
  const int n = idx / Kp;
  const int k = idx - n * Kp;
  o[idx] = (n < N && k < K) ? (_Float16)w[(size_t)n * K + k] : (_Float16)0.0f;
}

// ------------------- image f32 [B,300] -> f16 [B,320] padded ---------------
__global__ void pack_img_kernel(const float* __restrict__ src, _Float16* __restrict__ dst)
{
  const int idx = blockIdx.x * blockDim.x + threadIdx.x;
  if (idx >= B_ROWS * 320) return;
  const int r = idx / 320;
  const int c = idx - r * 320;
  dst[idx] = (c < 300) ? (_Float16)src[(size_t)r * 300 + c] : (_Float16)0.0f;
}

// --------------------- tag f32 [B,300] -> cols 10..309 ---------------------
__global__ void pack_tag_kernel(const float* __restrict__ tag, _Float16* __restrict__ dst)
{
  const int idx = blockIdx.x * blockDim.x + threadIdx.x;
  if (idx >= B_ROWS * 300) return;
  const int r = idx / 300;
  const int c = idx - r * 300;
  dst[(size_t)r * 320 + 10 + c] = (_Float16)tag[idx];
}

// --------- tiny loc MLP: 2->10->10->10, L2-normalize, NaN->0 ---------------
// One thread per row; writes cols 0..9 and zeros pad cols 310..319.
__global__ void loc_kernel(const float* __restrict__ lat, const float* __restrict__ lon,
                           const float* __restrict__ w1, const float* __restrict__ b1,
                           const float* __restrict__ w2, const float* __restrict__ b2,
                           const float* __restrict__ w3, const float* __restrict__ b3,
                           _Float16* __restrict__ dst /* [B,320] */)
{
  const int r = blockIdx.x * blockDim.x + threadIdx.x;
  if (r >= B_ROWS) return;
  const float la = lat[r], lo = lon[r];
  float h1[10], h2[10], h3[10];
#pragma unroll
  for (int j = 0; j < 10; ++j)
    h1[j] = fmaxf(w1[j * 2 + 0] * la + w1[j * 2 + 1] * lo + b1[j], 0.0f);
#pragma unroll
  for (int j = 0; j < 10; ++j) {
    float s = b2[j];
#pragma unroll
    for (int i = 0; i < 10; ++i) s += w2[j * 10 + i] * h1[i];
    h2[j] = fmaxf(s, 0.0f);
  }
  float nrm = 0.0f;
#pragma unroll
  for (int j = 0; j < 10; ++j) {
    float s = b3[j];
#pragma unroll
    for (int i = 0; i < 10; ++i) s += w3[j * 10 + i] * h2[i];
    h3[j] = fmaxf(s, 0.0f);
    nrm += h3[j] * h3[j];
  }
  nrm = sqrtf(nrm);
  _Float16* row = dst + (size_t)r * 320;
#pragma unroll
  for (int j = 0; j < 10; ++j) {
    float v = h3[j] / nrm;
    if (!(v == v)) v = 0.0f;   // loc[loc != loc] = 0
    row[j] = (_Float16)v;
  }
#pragma unroll
  for (int c = 310; c < 320; ++c) row[c] = (_Float16)0.0f;
}

// -------------------- pairwise distances + margin count --------------------
__global__ void zero_cnt_kernel(float* __restrict__ cnt) { *cnt = 0.0f; }

__global__ __launch_bounds__(256)
void dist_kernel(const float* __restrict__ outbuf, float* __restrict__ cnt)
{
  const int lane = threadIdx.x & 31;
  const int row  = (int)((blockIdx.x * blockDim.x + threadIdx.x) >> 5);
  if (row >= B_ROWS) return;
  const float* p = outbuf + (size_t)row * 300;                       // img_p_o
  const float* n = outbuf + (size_t)2457600 + (size_t)row * 300;     // img_n_o
  const float* a = outbuf + (size_t)4915200 + (size_t)row * 300;     // anchor
  float dp = 0.0f, dn = 0.0f;
  for (int c = lane; c < 300; c += 32) {
    const float av = a[c];
    const float d1 = fabsf(av - p[c] + 1e-6f);
    const float d2 = fabsf(av - n[c] + 1e-6f);
    dp += d1 * d1;
    dn += d2 * d2;
  }
#pragma unroll
  for (int m = 16; m >= 1; m >>= 1) {
    dp += __shfl_xor(dp, m, 32);
    dn += __shfl_xor(dn, m, 32);
  }
  if (lane == 0 && (sqrtf(dn) - sqrtf(dp)) > 0.1f) atomicAdd(cnt, 1.0f);
}

// ---------------------------------------------------------------------------
// Host-side orchestration.
// Input indices follow jax pytree flattening of setup_inputs(): top-level dict
// in insertion order, params dict leaves sorted alphabetically (b, gb, gw, w).
// ---------------------------------------------------------------------------
enum {
  IN_IMG_P = 0, IN_IMG_N = 1, IN_TAG = 2, IN_LAT = 3, IN_LON = 4,
  I1_B = 5,  I1_GB = 6,  I1_GW = 7,  I1_W = 8,
  I2_B = 9,  I2_GB = 10, I2_GW = 11, I2_W = 12,
  I3_B = 13, I3_GB = 14, I3_GW = 15, I3_W = 16,
  I4_B = 17, I4_W = 18,
  L1_B = 19, L1_W = 20, L2_B = 21, L2_W = 22, L3_B = 23, L3_W = 24,
  MM1_B = 25, MM1_GB = 26, MM1_GW = 27, MM1_W = 28,
  MM2_B = 29, MM2_GB = 30, MM2_GW = 31, MM2_W = 32,
  MM3_B = 33, MM3_GB = 34, MM3_GW = 35, MM3_W = 36,
  MM4_B = 37, MM4_GB = 38, MM4_GW = 39, MM4_W = 40,
  MM5_B = 41, MM5_W = 42,
};

extern "C" void kernel_launch(void* const* d_in, const int* in_sizes, int n_in,
                              void* d_out, int out_size, void* d_ws, size_t ws_size,
                              hipStream_t stream) {
  (void)in_sizes; (void)n_in; (void)out_size; (void)ws_size;
  auto fin = [&](int i) { return (const float*)d_in[i]; };

  // ---- workspace layout (in halves for f16, then f32 scratch) ----
  _Float16* ws16 = (_Float16*)d_ws;
  size_t o = 0;
  const size_t W_MM1 = o; o += (size_t)1024 * 320;
  const size_t W_MM2 = o; o += (size_t)2048 * 1024;
  const size_t W_MM3 = o; o += (size_t)2048 * 2048;
  const size_t W_MM4 = o; o += (size_t)1024 * 2048;
  const size_t W_MM5 = o; o += (size_t)384 * 1024;   // Np padded to 3*128
  const size_t W_I1  = o; o += (size_t)1024 * 320;
  const size_t W_I2  = o; o += (size_t)2048 * 1024;
  const size_t W_I3  = o; o += (size_t)1024 * 2048;
  const size_t W_I4  = o; o += (size_t)384 * 1024;   // Np padded to 3*128
  const size_t ACT_A = o; o += (size_t)B_ROWS * 2048;
  const size_t ACT_B = o; o += (size_t)B_ROWS * 2048;
  size_t f32_off_bytes = ((o * 2) + 255) & ~(size_t)255;
  float* f32buf = (float*)((char*)d_ws + f32_off_bytes);

  float* outf = (float*)d_out;
  float* out_img_p  = outf + 0;
  float* out_img_n  = outf + 2457600;
  float* out_anchor = outf + 4915200;
  float* out_cnt    = outf + 7372800;

  // ---- convert all weights to padded f16 ----
  struct WCvt { int idx; int N, K, Np, Kp; size_t dst; };
  const WCvt wc[9] = {
    { MM1_W, 1024,  310, 1024,  320, W_MM1 },
    { MM2_W, 2048, 1024, 2048, 1024, W_MM2 },
    { MM3_W, 2048, 2048, 2048, 2048, W_MM3 },
    { MM4_W, 1024, 2048, 1024, 2048, W_MM4 },
    { MM5_W,  300, 1024,  384, 1024, W_MM5 },
    { I1_W,  1024,  300, 1024,  320, W_I1  },
    { I2_W,  2048, 1024, 2048, 1024, W_I2  },
    { I3_W,  1024, 2048, 1024, 2048, W_I3  },
    { I4_W,   300, 1024,  384, 1024, W_I4  },
  };
  for (int i = 0; i < 9; ++i) {
    const int tot = wc[i].Np * wc[i].Kp;
    cvt_w_kernel<<<(tot + 255) / 256, 256, 0, stream>>>(
        fin(wc[i].idx), ws16 + wc[i].dst, wc[i].N, wc[i].K, wc[i].Np, wc[i].Kp);
  }

  auto gemm = [&](size_t actOff, size_t wOff, int biasIdx, float* out,
                  int Kp, int out_stride, int Nlim, int Npad) {
    dim3 grid(B_ROWS / 128, Npad / 128);
    gemm_wmma_kernel<<<grid, 256, 0, stream>>>(
        ws16 + actOff, ws16 + wOff, fin(biasIdx), out, Kp, out_stride, Nlim);
  };
  auto gn = [&](int gwIdx, int gbIdx, size_t dstOff, int C) {
    gn_relu_cvt_kernel<<<(B_ROWS * 32) / 8, 256, 0, stream>>>(
        f32buf, fin(gwIdx), fin(gbIdx), ws16 + dstOff, C);
  };

  // ======================= anchor (mm_net) =======================
  pack_tag_kernel<<<(B_ROWS * 300 + 255) / 256, 256, 0, stream>>>(
      fin(IN_TAG), ws16 + ACT_A);
  loc_kernel<<<(B_ROWS + 255) / 256, 256, 0, stream>>>(
      fin(IN_LAT), fin(IN_LON),
      fin(L1_W), fin(L1_B), fin(L2_W), fin(L2_B), fin(L3_W), fin(L3_B),
      ws16 + ACT_A);

  gemm(ACT_A, W_MM1, MM1_B, f32buf,  320, 1024, 1024, 1024);
  gn(MM1_GW, MM1_GB, ACT_B, 1024);
  gemm(ACT_B, W_MM2, MM2_B, f32buf, 1024, 2048, 2048, 2048);
  gn(MM2_GW, MM2_GB, ACT_A, 2048);
  gemm(ACT_A, W_MM3, MM3_B, f32buf, 2048, 2048, 2048, 2048);
  gn(MM3_GW, MM3_GB, ACT_B, 2048);
  gemm(ACT_B, W_MM4, MM4_B, f32buf, 2048, 1024, 1024, 1024);
  gn(MM4_GW, MM4_GB, ACT_A, 1024);
  gemm(ACT_A, W_MM5, MM5_B, out_anchor, 1024, 300, 300, 384);

  // ======================= img_p (img_net) =======================
  pack_img_kernel<<<(B_ROWS * 320 + 255) / 256, 256, 0, stream>>>(
      fin(IN_IMG_P), ws16 + ACT_A);
  gemm(ACT_A, W_I1, I1_B, f32buf,  320, 1024, 1024, 1024);
  gn(I1_GW, I1_GB, ACT_B, 1024);
  gemm(ACT_B, W_I2, I2_B, f32buf, 1024, 2048, 2048, 2048);
  gn(I2_GW, I2_GB, ACT_A, 2048);
  gemm(ACT_A, W_I3, I3_B, f32buf, 2048, 1024, 1024, 1024);
  gn(I3_GW, I3_GB, ACT_B, 1024);
  gemm(ACT_B, W_I4, I4_B, out_img_p, 1024, 300, 300, 384);

  // ======================= img_n (img_net) =======================
  pack_img_kernel<<<(B_ROWS * 320 + 255) / 256, 256, 0, stream>>>(
      fin(IN_IMG_N), ws16 + ACT_A);
  gemm(ACT_A, W_I1, I1_B, f32buf,  320, 1024, 1024, 1024);
  gn(I1_GW, I1_GB, ACT_B, 1024);
  gemm(ACT_B, W_I2, I2_B, f32buf, 1024, 2048, 2048, 2048);
  gn(I2_GW, I2_GB, ACT_A, 2048);
  gemm(ACT_A, W_I3, I3_B, f32buf, 2048, 1024, 1024, 1024);
  gn(I3_GW, I3_GB, ACT_B, 1024);
  gemm(ACT_B, W_I4, I4_B, out_img_n, 1024, 300, 300, 384);

  // =============== distances + margin count ===============
  zero_cnt_kernel<<<1, 1, 0, stream>>>(out_cnt);
  dist_kernel<<<(B_ROWS * 32) / 256, 256, 0, stream>>>(outf, out_cnt);
}
